// IENModel_59528246723084
// MI455X (gfx1250) — compile-verified, matrix operand order checked by reading, not run on hardware
//
#include <hip/hip_runtime.h>
#include <hip/hip_bf16.h>

// ---------------------------------------------------------------------------
// CDNA5 (gfx1250) implementation: wave32, v_wmma_f32_16x16x32_f16 for all
// GEMMs, global_load_async_to_lds_b128 staging (ASYNCcnt), cluster barrier
// for the serial LSTM recurrence.
// ---------------------------------------------------------------------------

typedef __attribute__((ext_vector_type(16))) _Float16 v16h;
typedef __attribute__((ext_vector_type(8)))  float    v8f;

#define W_LEN 4096
#define SSEG  128
#define DEMB  300
#define HH    512
#define EE    64
#define LL    64
#define TT    130   // S+2
#define BB    128   // E+L

__device__ __forceinline__ float sigm(float x) { return 1.f / (1.f + __expf(-x)); }

// Async 16B copy global -> LDS (gfx1250, tracked by ASYNCcnt). The low 32 bits
// of a generic pointer into LDS are the LDS byte offset (aperture mapping).
__device__ __forceinline__ void async_ld_b128(const void* gptr, void* lptr) {
  unsigned lds = (unsigned)(size_t)lptr;
  asm volatile("global_load_async_to_lds_b128 %0, %1, off"
               :: "v"(lds), "v"(gptr) : "memory");
}
__device__ __forceinline__ void wait_async0() {
  asm volatile("s_wait_asynccnt 0" ::: "memory");
}

// Load one 16x32 (rows x K) f16 fragment from an LDS tile stored row-major
// with leading dim `ld`, per the CDNA5 16-bit A-matrix layout (ISA 7.12.2).
// The B operand (K x 16) uses the mirrored layout, which is the same gather
// when the weight tile is staged as its native (N,K) rows.
__device__ __forceinline__ v16h frag_rowK(const _Float16* base, int ld) {
  int lane = threadIdx.x & 31;
  int lo = lane & 15;
  int hi = (lane >> 4) & 1;
  const _Float16* row = base + lo * ld;
  v16h f;
#pragma unroll
  for (int i = 0; i < 16; ++i) {
    int k = (i < 8) ? (hi * 8 + i) : (8 + hi * 8 + i);
    f[i] = row[k];
  }
  return f;
}

#define WMMA_F16(A_, B_, C_) \
  __builtin_amdgcn_wmma_f32_16x16x32_f16(false, (A_), false, (B_), (short)0, (C_), false, false)

// ---------------------------------------------------------------------------
// Generic C[M][N] = A[M][K] * W[N][K]^T, f16 inputs, f32 out.
// Block: 256 threads (8 waves). Tile: 128(M) x 64(N), K step 32.
// Wave w owns a 2x2 grid of 16x16 tiles: m-tiles {2*(w>>1), +1}, n-tiles
// {2*(w&1), +1} -> 4 WMMA per K-step with A/B fragment reuse.
// Requires M%128==0, N%64==0, K%32==0 (true for every call site here).
// ---------------------------------------------------------------------------
__global__ __launch_bounds__(256) void k_gemm(const _Float16* __restrict__ A,
                                              const _Float16* __restrict__ W,
                                              float* __restrict__ C,
                                              int M, int N, int K) {
  __shared__ _Float16 sA[128 * 32];
  __shared__ _Float16 sW[64 * 32];
  int tid  = threadIdx.x;
  int wave = tid >> 5;
  int lane = tid & 31;
  int mGroup = wave >> 1;       // 0..3
  int nGroup = wave & 1;        // 0..1
  int mBase = blockIdx.x * 128;
  int nBase = blockIdx.y * 64;
  v8f acc00 = {}, acc01 = {}, acc10 = {}, acc11 = {};
  for (int k0 = 0; k0 < K; k0 += 32) {
    // A tile: 128x32 halfs = 512 x 16B chunks; 2 per thread (async -> LDS)
#pragma unroll
    for (int it = 0; it < 2; ++it) {
      int c   = tid + it * 256;           // 0..511
      int row = c >> 2;
      int col = (c & 3) * 8;
      async_ld_b128(A + (size_t)(mBase + row) * K + k0 + col, &sA[row * 32 + col]);
    }
    // W tile: 64x32 halfs = 256 x 16B chunks; 1 per thread
    {
      int row = tid >> 2;
      int col = (tid & 3) * 8;
      async_ld_b128(W + (size_t)(nBase + row) * K + k0 + col, &sW[row * 32 + col]);
    }
    wait_async0();
    __syncthreads();
    v16h a0 = frag_rowK(&sA[(mGroup * 2 + 0) * 16 * 32], 32);
    v16h a1 = frag_rowK(&sA[(mGroup * 2 + 1) * 16 * 32], 32);
    v16h b0 = frag_rowK(&sW[(nGroup * 2 + 0) * 16 * 32], 32);
    v16h b1 = frag_rowK(&sW[(nGroup * 2 + 1) * 16 * 32], 32);
    acc00 = WMMA_F16(a0, b0, acc00);
    acc01 = WMMA_F16(a0, b1, acc01);
    acc10 = WMMA_F16(a1, b0, acc10);
    acc11 = WMMA_F16(a1, b1, acc11);
    __syncthreads();
  }
  int lo = lane & 15, hi = (lane >> 4) & 1;
#pragma unroll
  for (int r = 0; r < 8; ++r) {
    int m0 = mBase + (mGroup * 2 + 0) * 16 + hi * 8 + r;
    int m1 = mBase + (mGroup * 2 + 1) * 16 + hi * 8 + r;
    int n0 = nBase + (nGroup * 2 + 0) * 16 + lo;
    int n1 = n0 + 16;
    C[(size_t)m0 * N + n0] = acc00[r];
    C[(size_t)m0 * N + n1] = acc01[r];
    C[(size_t)m1 * N + n0] = acc10[r];
    C[(size_t)m1 * N + n1] = acc11[r];
  }
}

// ---------------------------------------------------------------------------
// Embedding gather + verb concat, padded to K=320, f16.
// ---------------------------------------------------------------------------
__global__ void k_embed(const int* __restrict__ idxs, const float* __restrict__ emb,
                        const float* __restrict__ verbs, _Float16* __restrict__ x) {
  int t = blockIdx.x;
  int w = idxs[t];
  _Float16* row = x + (size_t)t * 320;
  const float* e = emb + (size_t)w * DEMB;
  for (int d = threadIdx.x; d < 320; d += blockDim.x) {
    float v = (d < DEMB) ? e[d] : ((d == DEMB) ? verbs[t] : 0.f);
    row[d] = (_Float16)v;
  }
}

// f32 (rows,srcK) -> f16 (rows,dstK) zero-padded
__global__ void k_cvt(const float* __restrict__ src, _Float16* __restrict__ dst,
                      int rows, int srcK, int dstK) {
  size_t i = (size_t)blockIdx.x * blockDim.x + threadIdx.x;
  size_t total = (size_t)rows * dstK;
  if (i >= total) return;
  int r = (int)(i / dstK), k = (int)(i % dstK);
  dst[i] = (_Float16)((k < srcK) ? src[(size_t)r * srcK + k] : 0.f);
}

// ---------------------------------------------------------------------------
// LSTM recurrence. 16 workgroups form one cluster per direction; workgroup b
// owns gate rows {g*512 + 32b .. +32} for g in {i,f,g,o} and hidden slice
// [32b,32b+32). h is shared through L2; steps are separated by the CDNA5
// cluster barrier (wave 0 signals+waits, NOP-degrades if not clustered).
// ---------------------------------------------------------------------------
__global__ __launch_bounds__(128) void k_lstm(const float* __restrict__ pre,   // [T][2048]
                                              const float* __restrict__ Whh,   // [2048][512] f32
                                              const float* __restrict__ bias,  // [2048]
                                              float* __restrict__ hbuf,        // [512] shared
                                              float* __restrict__ enc,         // [T][1024]
                                              int dir) {
  int b = blockIdx.x;            // 0..15
  int tid = threadIdx.x;         // 0..127
  __shared__ float sh[512];
  __shared__ float sg[128];
  float c = 0.f;                 // cell state, owned by tid<32
  if (tid < 32) hbuf[b * 32 + tid] = 0.f;
  __threadfence();
  __syncthreads();
  if (tid < 32) __builtin_amdgcn_s_cluster_barrier();
  __syncthreads();
  for (int s = 0; s < W_LEN; ++s) {
    int t = dir ? (W_LEN - 1 - s) : s;
    for (int i = tid; i < 512; i += 128) sh[i] = hbuf[i];
    __syncthreads();
    int grp = tid >> 5, jl = tid & 31;
    int grow = grp * 512 + b * 32 + jl;
    const float* wr = Whh + (size_t)grow * 512;
    float acc = pre[(size_t)t * 2048 + grow] + bias[grow];
#pragma unroll 4
    for (int k = 0; k < 512; ++k) acc = fmaf(wr[k], sh[k], acc);
    sg[tid] = acc;
    __syncthreads();
    if (tid < 32) {
      float ig = sg[tid], fg = sg[32 + tid], gg = sg[64 + tid], og = sg[96 + tid];
      c = sigm(fg) * c + sigm(ig) * tanhf(gg);
      float h = sigm(og) * tanhf(c);
      hbuf[b * 32 + tid] = h;
      enc[(size_t)t * 1024 + dir * 512 + b * 32 + tid] = h;
    }
    __threadfence();
    __syncthreads();
    if (tid < 32) __builtin_amdgcn_s_cluster_barrier();
    __syncthreads();
  }
}

// ---------------------------------------------------------------------------
// Verb segment sums / mean, prefix sums, span features.
// ---------------------------------------------------------------------------
__global__ void k_zero_f(float* p, int n) {
  int i = blockIdx.x * blockDim.x + threadIdx.x;
  if (i < n) p[i] = 0.f;
}
__global__ void k_zero_h(_Float16* p, int n) {
  int i = blockIdx.x * blockDim.x + threadIdx.x;
  if (i < n) p[i] = (_Float16)0.f;
}

__global__ void k_vsum(const float* __restrict__ enc, const float* __restrict__ verbs,
                       const int* __restrict__ sent_ids,
                       float* __restrict__ vsum, float* __restrict__ vcnt) {
  int t = blockIdx.x;
  if (!(verbs[t] > 0.5f)) return;
  int sid = sent_ids[t];
  for (int d = threadIdx.x; d < 1024; d += blockDim.x)
    atomicAdd(&vsum[(size_t)sid * 1024 + d], enc[(size_t)t * 1024 + d]);
  if (threadIdx.x == 0) atomicAdd(&vcnt[sid], 1.f);
}

__global__ void k_vmean(const float* __restrict__ vsum, const float* __restrict__ vcnt,
                        float* __restrict__ vmean) {
  int i = blockIdx.x * blockDim.x + threadIdx.x;
  if (i >= SSEG * 1024) return;
  float c = vcnt[i >> 10];
  vmean[i] = (c > 0.f) ? vsum[i] / fmaxf(c, 1.f) : 0.f;
}

__global__ void k_prefix(const float* __restrict__ enc, float* __restrict__ pref) {
  int d = blockIdx.x * blockDim.x + threadIdx.x;
  if (d >= 1024) return;
  float run = 0.f;
  pref[d] = 0.f;
  for (int t = 0; t < W_LEN; ++t) {
    run += enc[(size_t)t * 1024 + d];
    pref[(size_t)(t + 1) * 1024 + d] = run;
  }
}

// Span features -> sl_f16 [130][128][2048]. grid (130, 64) per span family.
// Reproduces the reference's verbs_padded[:S] shift (vpart row s == s==0?0:vmean[s-1]).
__global__ void k_span(const int* __restrict__ startArr, const int* __restrict__ lenArr,
                       const int* __restrict__ maskArr, const float* __restrict__ pref,
                       const float* __restrict__ vmean, _Float16* __restrict__ sl,
                       int batchOffset) {
  int s = blockIdx.x;            // 0..129 (padded time)
  int e = blockIdx.y;            // 0..63
  _Float16* out = sl + (((size_t)s * BB) + batchOffset + e) * 2048;
  if (s == 0 || s == TT - 1) {
    for (int d = threadIdx.x; d < 2048; d += blockDim.x) out[d] = (_Float16)0.f;
    return;
  }
  int sent = s - 1;
  int st = startArr[e * SSEG + sent];
  int ln = lenArr[e * SSEG + sent];
  float mk = (float)maskArr[e * SSEG + sent];
  int end = st + ln; if (end > W_LEN - 1) end = W_LEN - 1;
  float inv = 1.f / (float)(end - st + 1);
  for (int d = threadIdx.x; d < 2048; d += blockDim.x) {
    float v;
    if (d < 1024)
      v = (pref[(size_t)(end + 1) * 1024 + d] - pref[(size_t)st * 1024 + d]) * inv;
    else
      v = (sent == 0) ? 0.f : vmean[(size_t)(sent - 1) * 1024 + (d - 1024)];
    out[d] = (_Float16)(v * mk);
  }
}

// ---------------------------------------------------------------------------
// GRU step elementwise update (gi precomputed by WMMA GEMM, gh per step).
// ---------------------------------------------------------------------------
__global__ void k_gru_update(const float* __restrict__ gi, const float* __restrict__ gh,
                             const float* __restrict__ bias,
                             float* __restrict__ h, _Float16* __restrict__ h16,
                             float* __restrict__ rnn, int t, int dir) {
  int idx = blockIdx.x * blockDim.x + threadIdx.x;
  if (idx >= BB * HH) return;
  int bb = idx >> 9, j = idx & 511;
  const float* gir = gi + ((size_t)t * BB + bb) * 1536;
  const float* ghr = gh + (size_t)bb * 1536;
  float r = sigm(gir[j] + bias[j] + ghr[j]);
  float z = sigm(gir[512 + j] + bias[512 + j] + ghr[512 + j]);
  float m = tanhf(gir[1024 + j] + bias[1024 + j] + r * ghr[1024 + j]);
  float hv = (1.f - z) * m + z * h[idx];
  h[idx] = hv;
  h16[idx] = (_Float16)hv;
  rnn[((size_t)t * BB + bb) * 1024 + dir * 512 + j] = hv;
}

// rnn (130,128,1024) f32 -> ent rows / loc rows f16
__global__ void k_split_rnn(const float* __restrict__ rnn,
                            _Float16* __restrict__ entA, _Float16* __restrict__ locA) {
  int row = blockIdx.x;                  // 0..16639
  int s = row >> 7, bb = row & 127;
  const float* src = rnn + (size_t)row * 1024;
  _Float16* dst = (bb < EE) ? entA + ((size_t)s * EE + bb) * 1024
                            : locA + ((size_t)s * LL + (bb - EE)) * 1024;
  for (int d = threadIdx.x; d < 1024; d += blockDim.x) dst[d] = (_Float16)src[d];
}

// LayerNorm(bias-added) + ReLU over last dim 512; optional f32/f16 outputs.
__global__ __launch_bounds__(256) void k_ln_relu(const float* __restrict__ in,
                                                 const float* __restrict__ bias,
                                                 const float* __restrict__ g,
                                                 const float* __restrict__ bta,
                                                 float* __restrict__ out32,
                                                 _Float16* __restrict__ out16) {
  int row = blockIdx.x;
  const float* x = in + (size_t)row * 512;
  __shared__ float red[256];
  int tid = threadIdx.x;
  float s = 0.f;
  for (int d = tid; d < 512; d += 256) s += x[d] + bias[d];
  red[tid] = s; __syncthreads();
  for (int off = 128; off; off >>= 1) { if (tid < off) red[tid] += red[tid + off]; __syncthreads(); }
  float mu = red[0] / 512.f;
  __syncthreads();
  float v = 0.f;
  for (int d = tid; d < 512; d += 256) { float tt = x[d] + bias[d] - mu; v += tt * tt; }
  red[tid] = v; __syncthreads();
  for (int off = 128; off; off >>= 1) { if (tid < off) red[tid] += red[tid + off]; __syncthreads(); }
  float rstd = rsqrtf(red[0] / 512.f + 1e-5f);
  for (int d = tid; d < 512; d += 256) {
    float y = (x[d] + bias[d] - mu) * rstd * g[d] + bta[d];
    y = fmaxf(y, 0.f);
    if (out32) out32[(size_t)row * 512 + d] = y;
    if (out16) out16[(size_t)row * 512 + d] = (_Float16)y;
  }
}

// states_logit = s1 @ W2^T + b2 (N=6, tiny -> VALU)
__global__ void k_state2(const float* __restrict__ s1, const float* __restrict__ W2,
                         const float* __restrict__ b2, float* __restrict__ out) {
  int i = blockIdx.x * blockDim.x + threadIdx.x;
  if (i >= TT * EE * 6) return;
  int row = i / 6, cc = i % 6;
  const float* x = s1 + (size_t)row * 512;
  const float* w = W2 + (size_t)cc * 512;
  float acc = b2[cc];
  for (int k = 0; k < 512; ++k) acc = fmaf(x[k], w[k], acc);
  out[i] = acc;
}

// rel[s] = es_s (64x512) @ el_s^T -> 64x64, WMMA per sentence block.
__global__ __launch_bounds__(256) void k_rel(const _Float16* __restrict__ ES,
                                             const _Float16* __restrict__ EL,
                                             float* __restrict__ OUT) {
  __shared__ _Float16 sA[64 * 32];
  __shared__ _Float16 sB[64 * 32];
  int s = blockIdx.x;
  const _Float16* Aes = ES + (size_t)s * 64 * 512;
  const _Float16* Bel = EL + (size_t)s * 64 * 512;
  int tid = threadIdx.x, wave = tid >> 5, lane = tid & 31;
  int mt = wave >> 1, ntB = (wave & 1) * 2;
  v8f acc0 = {}, acc1 = {};
  for (int k0 = 0; k0 < 512; k0 += 32) {
    int row = tid >> 2, col = (tid & 3) * 8;
    async_ld_b128(Aes + (size_t)row * 512 + k0 + col, &sA[row * 32 + col]);
    async_ld_b128(Bel + (size_t)row * 512 + k0 + col, &sB[row * 32 + col]);
    wait_async0();
    __syncthreads();
    v16h a  = frag_rowK(&sA[mt * 16 * 32], 32);
    v16h b0 = frag_rowK(&sB[ntB * 16 * 32], 32);
    v16h b1 = frag_rowK(&sB[(ntB + 1) * 16 * 32], 32);
    acc0 = WMMA_F16(a, b0, acc0);
    acc1 = WMMA_F16(a, b1, acc1);
    __syncthreads();
  }
  int lo = lane & 15, hi = (lane >> 4) & 1;
  float* o = OUT + (size_t)s * 64 * 64;
#pragma unroll
  for (int r = 0; r < 8; ++r) {
    int m = mt * 16 + hi * 8 + r;
    o[m * 64 + ntB * 16 + lo]       = acc0[r];
    o[m * 64 + (ntB + 1) * 16 + lo] = acc1[r];
  }
}

// ---------------------------------------------------------------------------
// Host orchestration
// ---------------------------------------------------------------------------
static inline char* bump(char*& p, size_t bytes) {
  char* r = p;
  p += (bytes + 255) & ~(size_t)255;
  return r;
}

extern "C" void kernel_launch(void* const* d_in, const int* in_sizes, int n_in,
                              void* d_out, int out_size, void* d_ws, size_t ws_size,
                              hipStream_t stream) {
  (void)in_sizes; (void)n_in; (void)out_size; (void)ws_size;

  const int*   words    = (const int*)d_in[0];
  const float* verbs    = (const float*)d_in[1];
  const int*   sent_ids = (const int*)d_in[2];
  const int*   ent_st   = (const int*)d_in[3];
  const int*   ent_ln   = (const int*)d_in[4];
  const int*   ent_mk   = (const int*)d_in[5];
  const int*   loc_st   = (const int*)d_in[6];
  const int*   loc_ln   = (const int*)d_in[7];
  const int*   loc_mk   = (const int*)d_in[8];
  const float* embedding= (const float*)d_in[9];
  const float* lf_Wih = (const float*)d_in[10]; const float* lf_Whh = (const float*)d_in[11];
  const float* lf_b   = (const float*)d_in[12];
  const float* gf_Wih = (const float*)d_in[13]; const float* gf_Whh = (const float*)d_in[14];
  const float* gf_b   = (const float*)d_in[15];
  const float* lb_Wih = (const float*)d_in[16]; const float* lb_Whh = (const float*)d_in[17];
  const float* lb_b   = (const float*)d_in[18];
  const float* gb_Wih = (const float*)d_in[19]; const float* gb_Whh = (const float*)d_in[20];
  const float* gb_b   = (const float*)d_in[21];
  const float* state_W1 = (const float*)d_in[22]; const float* state_b1 = (const float*)d_in[23];
  const float* state_g  = (const float*)d_in[24]; const float* state_bt = (const float*)d_in[25];
  const float* state_W2 = (const float*)d_in[26]; const float* state_b2 = (const float*)d_in[27];
  const float* loc1_W = (const float*)d_in[28]; const float* loc1_b = (const float*)d_in[29];
  const float* loc1_g = (const float*)d_in[30]; const float* loc1_bt = (const float*)d_in[31];
  const float* loc2_W = (const float*)d_in[32]; const float* loc2_b = (const float*)d_in[33];
  const float* loc2_g = (const float*)d_in[34]; const float* loc2_bt = (const float*)d_in[35];

  float* out_states = (float*)d_out;                         // 130*64*6
  float* out_rel    = (float*)d_out + (size_t)TT * EE * 6;   // 130*64*64

  char* p = (char*)d_ws;
  _Float16* x16    = (_Float16*)bump(p, (size_t)W_LEN * 320 * 2);
  _Float16* wih16  = (_Float16*)bump(p, (size_t)2048 * 320 * 2);       // reused f/b
  float*    pre    = (float*)bump(p, (size_t)W_LEN * 2048 * 4);        // reused f/b
  float*    enc    = (float*)bump(p, (size_t)W_LEN * 1024 * 4);
  float*    hls    = (float*)bump(p, 512 * 4);
  float*    vsum   = (float*)bump(p, (size_t)SSEG * 1024 * 4);
  float*    vcnt   = (float*)bump(p, SSEG * 4);
  float*    vmean  = (float*)bump(p, (size_t)SSEG * 1024 * 4);
  float*    pref   = (float*)bump(p, (size_t)(W_LEN + 1) * 1024 * 4);
  _Float16* sl16   = (_Float16*)bump(p, (size_t)TT * BB * 2048 * 2);
  _Float16* gwih16 = (_Float16*)bump(p, (size_t)1536 * 2048 * 2);      // reused f/b
  _Float16* gwhh16 = (_Float16*)bump(p, (size_t)1536 * 512 * 2);       // reused f/b
  float*    gi     = (float*)bump(p, (size_t)TT * BB * 1536 * 4);      // reused f/b
  float*    gh     = (float*)bump(p, (size_t)BB * 1536 * 4);
  float*    hg     = (float*)bump(p, (size_t)BB * HH * 4);
  _Float16* hg16   = (_Float16*)bump(p, (size_t)BB * HH * 2);
  float*    rnn    = (float*)bump(p, (size_t)TT * BB * 1024 * 4);
  _Float16* entA   = (_Float16*)bump(p, (size_t)TT * EE * 1024 * 2);
  _Float16* locA   = (_Float16*)bump(p, (size_t)TT * LL * 1024 * 2);
  _Float16* w16h   = (_Float16*)bump(p, (size_t)512 * 1024 * 2);       // reused x3
  float*    tbuf   = (float*)bump(p, (size_t)TT * EE * 512 * 4);       // reused x3
  float*    s1     = (float*)bump(p, (size_t)TT * EE * 512 * 4);
  _Float16* es16   = (_Float16*)bump(p, (size_t)TT * EE * 512 * 2);
  _Float16* el16   = (_Float16*)bump(p, (size_t)TT * LL * 512 * 2);

  // ---- LSTM input transform + recurrence (forward then backward) ----
  k_embed<<<W_LEN, 64, 0, stream>>>(words, embedding, verbs, x16);

  { int tot = 2048 * 320;
    k_cvt<<<(tot + 255) / 256, 256, 0, stream>>>(lf_Wih, wih16, 2048, DEMB + 1, 320);
    dim3 g(W_LEN / 128, 2048 / 64);
    k_gemm<<<g, 256, 0, stream>>>(x16, wih16, pre, W_LEN, 2048, 320);
    k_lstm<<<16, 128, 0, stream>>>(pre, lf_Whh, lf_b, hls, enc, 0);
    k_cvt<<<(tot + 255) / 256, 256, 0, stream>>>(lb_Wih, wih16, 2048, DEMB + 1, 320);
    k_gemm<<<g, 256, 0, stream>>>(x16, wih16, pre, W_LEN, 2048, 320);
    k_lstm<<<16, 128, 0, stream>>>(pre, lb_Whh, lb_b, hls, enc, 1);
  }

  // ---- verb means + prefix sums + span features ----
  k_zero_f<<<(SSEG * 1024 + 255) / 256, 256, 0, stream>>>(vsum, SSEG * 1024);
  k_zero_f<<<1, 128, 0, stream>>>(vcnt, SSEG);
  k_vsum<<<W_LEN, 256, 0, stream>>>(enc, verbs, sent_ids, vsum, vcnt);
  k_vmean<<<(SSEG * 1024 + 255) / 256, 256, 0, stream>>>(vsum, vcnt, vmean);
  k_prefix<<<4, 256, 0, stream>>>(enc, pref);
  { dim3 g(TT, EE);
    k_span<<<g, 256, 0, stream>>>(ent_st, ent_ln, ent_mk, pref, vmean, sl16, 0);
    k_span<<<g, 256, 0, stream>>>(loc_st, loc_ln, loc_mk, pref, vmean, sl16, EE);
  }

  // ---- GRU: big WMMA input GEMM + 130-step recurrence per direction ----
  for (int dir = 0; dir < 2; ++dir) {
    const float* Wih = dir ? gb_Wih : gf_Wih;
    const float* Whh = dir ? gb_Whh : gf_Whh;
    const float* bia = dir ? gb_b   : gf_b;
    { int tot = 1536 * 2048;
      k_cvt<<<(tot + 255) / 256, 256, 0, stream>>>(Wih, gwih16, 1536, 2048, 2048); }
    { int tot = 1536 * 512;
      k_cvt<<<(tot + 255) / 256, 256, 0, stream>>>(Whh, gwhh16, 1536, 512, 512); }
    { dim3 g((TT * BB) / 128, 1536 / 64);
      k_gemm<<<g, 256, 0, stream>>>(sl16, gwih16, gi, TT * BB, 1536, 2048); }
    k_zero_f<<<(BB * HH + 255) / 256, 256, 0, stream>>>(hg, BB * HH);
    k_zero_h<<<(BB * HH + 255) / 256, 256, 0, stream>>>(hg16, BB * HH);
    for (int s = 0; s < TT; ++s) {
      int t = dir ? (TT - 1 - s) : s;
      dim3 g(1, 1536 / 64);
      k_gemm<<<g, 256, 0, stream>>>(hg16, gwhh16, gh, BB, 1536, 512);
      k_gru_update<<<(BB * HH + 255) / 256, 256, 0, stream>>>(gi, gh, bia, hg, hg16, rnn, t, dir);
    }
  }

  // ---- heads ----
  k_split_rnn<<<TT * BB, 256, 0, stream>>>(rnn, entA, locA);

  { int tot = 512 * 1024;
    dim3 g((TT * EE) / 128, 512 / 64);
    // state head
    k_cvt<<<(tot + 255) / 256, 256, 0, stream>>>(state_W1, w16h, 512, 1024, 1024);
    k_gemm<<<g, 256, 0, stream>>>(entA, w16h, tbuf, TT * EE, 512, 1024);
    k_ln_relu<<<TT * EE, 256, 0, stream>>>(tbuf, state_b1, state_g, state_bt, s1, (_Float16*)nullptr);
    k_state2<<<(TT * EE * 6 + 255) / 256, 256, 0, stream>>>(s1, state_W2, state_b2, out_states);
    // es = relu(LN(ent @ loc1_W^T + loc1_b))
    k_cvt<<<(tot + 255) / 256, 256, 0, stream>>>(loc1_W, w16h, 512, 1024, 1024);
    k_gemm<<<g, 256, 0, stream>>>(entA, w16h, tbuf, TT * EE, 512, 1024);
    k_ln_relu<<<TT * EE, 256, 0, stream>>>(tbuf, loc1_b, loc1_g, loc1_bt, (float*)nullptr, es16);
    // el = relu(LN(loc @ loc2_W^T + loc2_b))
    k_cvt<<<(tot + 255) / 256, 256, 0, stream>>>(loc2_W, w16h, 512, 1024, 1024);
    k_gemm<<<g, 256, 0, stream>>>(locA, w16h, tbuf, TT * LL, 512, 1024);
    k_ln_relu<<<TT * LL, 256, 0, stream>>>(tbuf, loc2_b, loc2_g, loc2_bt, (float*)nullptr, el16);
  }

  // rel = einsum('sed,sld->sel')
  k_rel<<<TT, 256, 0, stream>>>(es16, el16, out_rel);
}
